// SelfAttention_2637109920370
// MI455X (gfx1250) — matrix operand order repaired and checked by reading, not compiled, not run on hardware
//
#include <hip/hip_runtime.h>

// ---------------------------------------------------------------------------
// CDNA5 (gfx1250) bf16 WMMA self-attention pipeline.
//  - all matmuls on v_wmma_f32_16x16x32_bf16 (wave32, f32 accum)
//  - GEMM tiles staged into LDS via Tensor Data Mover (tensor_load_to_lds,
//    TENSORcnt double-buffering) when the builtin is available
// ---------------------------------------------------------------------------

typedef __attribute__((ext_vector_type(16))) __bf16 v16bf;
typedef __attribute__((ext_vector_type(8)))  float  v8f;
typedef __attribute__((ext_vector_type(4)))  unsigned int u32x4;
typedef __attribute__((ext_vector_type(8)))  int    i32x8;
typedef __attribute__((ext_vector_type(4)))  int    i32x4;

union Frag {
    v16bf v;
    u32x4 q[2];
    unsigned short u[16];
};

__device__ __forceinline__ unsigned short f2bf(float f) {
    unsigned int u = __float_as_uint(f);
    unsigned int r = (u + 0x7FFFu + ((u >> 16) & 1u)) >> 16;  // RNE
    return (unsigned short)r;
}

__device__ __forceinline__ v8f bwmma(const Frag& a, const Frag& b, v8f c) {
    // D = A(16x32 bf16) x B(32x16 bf16) + C(16x16 f32)
    return __builtin_amdgcn_wmma_f32_16x16x32_bf16(
        false, a.v, false, b.v, (short)0, c, false, false);
}

__device__ __forceinline__ u32x4 ld128(const unsigned short* p) {
    return *reinterpret_cast<const u32x4*>(p);
}

// A-fragment (16x32): lane L<16 -> row L, K {0..7, 16..23};
// lane L>=16 -> row L-16, K {8..15, 24..31}. base -> [row0][k0], ld in elems.
__device__ __forceinline__ void load_afrag(Frag& f, const unsigned short* base,
                                           int ld, int lane) {
    const unsigned short* p = base + (size_t)(lane & 15) * ld + ((lane >> 4) & 1) * 8;
    f.q[0] = ld128(p);
    f.q[1] = ld128(p + 16);
}

// B-fragment (32x16): lane L<16 -> col L, K {0..7, 16..23}; lane L>=16 -> col L-16,
// K {8..15, 24..31}. base -> [col0][k0] of the K-major source, ld in elems.
__device__ __forceinline__ void load_bfrag(Frag& f, const unsigned short* base,
                                           size_t ld, int lane) {
    const unsigned short* p = base + (size_t)(lane & 15) * ld + ((lane >> 4) & 1) * 8;
    f.q[0] = ld128(p);
    f.q[1] = ld128(p + 16);
}

// ---------------------------------------------------------------------------
// fp32 -> bf16 cast
// ---------------------------------------------------------------------------
__global__ void cast_bf16_kernel(const float* __restrict__ in,
                                 unsigned short* __restrict__ out, int n) {
    int i = blockIdx.x * blockDim.x + threadIdx.x;
    if (i < n) out[i] = f2bf(in[i]);
}

// ---------------------------------------------------------------------------
// TDM support
// ---------------------------------------------------------------------------
#if defined(__has_builtin)
#if __has_builtin(__builtin_amdgcn_tensor_load_to_lds) && \
    __has_builtin(__builtin_amdgcn_s_wait_tensorcnt)
#define USE_TDM 1
#endif
#endif
#ifndef USE_TDM
#define USE_TDM 0
#endif

#if USE_TDM
// Generic shared pointer low 32 bits == workgroup-relative LDS byte offset
// (ISA 10.2).
__device__ __forceinline__ unsigned lds_addr_of(const void* p) {
    return (unsigned)(unsigned long long)p;
}

// 2D TDM load: tile_rows x 32 bf16 elements, row pitch `row_stride` elems in
// global memory. LDS padding: +4 DWORDs (16B) after every 16 DWORDs (64B) so
// the LDS row pitch becomes 80B (40 elems) for conflict-reduced b128 reads.
__device__ __forceinline__ void tdm_load_tile(const void* gptr,
                                              unsigned lds_byte_off,
                                              unsigned tile_rows,
                                              unsigned tensor_k,
                                              unsigned tensor_rows,
                                              unsigned row_stride) {
    unsigned long long ga = (unsigned long long)gptr;
    u32x4 g0;
    g0[0] = 1u;                                   // count=1 (valid), user mode
    g0[1] = lds_byte_off;                         // lds_addr
    g0[2] = (unsigned)(ga & 0xFFFFFFFFu);         // global_addr[31:0]
    g0[3] = (unsigned)((ga >> 32) & 0x01FFFFFFu)  // global_addr[56:32]
            | (2u << 30);                         // type = 2 (image)
    i32x8 g1;
    g1[0] = (int)((1u << 16)                      // data_size = 1 (2 bytes)
                  | (1u << 20)                    // pad_enable
                  | (3u << 22)                    // pad_interval: 16 DWORDs
                  | (3u << 25));                  // pad_amount: 4 DWORDs
    g1[1] = (int)((tensor_k & 0xFFFFu) << 16);                     // dim0 lo
    g1[2] = (int)(((tensor_k >> 16) & 0xFFFFu) |
                  ((tensor_rows & 0xFFFFu) << 16));                // dim0 hi|dim1 lo
    g1[3] = (int)(((tensor_rows >> 16) & 0xFFFFu) | (32u << 16));  // dim1 hi|tile_dim0
    g1[4] = (int)(tile_rows & 0xFFFFu);                            // tile_dim1
    g1[5] = (int)row_stride;                                       // dim0_stride lo32
    g1[6] = 0;
    g1[7] = 0;
    i32x4 z4 = {0, 0, 0, 0};
#if __clang_major__ >= 23
    i32x8 z8 = {0, 0, 0, 0, 0, 0, 0, 0};
    __builtin_amdgcn_tensor_load_to_lds(g0, g1, z4, z4, z8, 0);
#else
    __builtin_amdgcn_tensor_load_to_lds(g0, g1, z4, z4, 0);
#endif
}

// ---------------------------------------------------------------------------
// TDM-staged GEMM: C = A[M,K] x W^T, W stored [N,K] row-major.
// MODE 0: C bf16 [M,N];  MODE 1: C bf16 transposed [N,M];  MODE 2: C f32 [M,N]
// Block 128 threads = 4 waves (each 64x64), block tile 128x128, k-step 32.
// Double-buffered LDS tiles filled by the Tensor Data Mover.
// ---------------------------------------------------------------------------
#define LDS_PITCH 40  // 32 elems + TDM pad (4 DWORDs per 16) -> 80B row pitch

template <int MODE>
__global__ __launch_bounds__(128) void gemm_bf16_kernel(
    const unsigned short* __restrict__ A,
    const unsigned short* __restrict__ W,
    void* __restrict__ Cout,
    int M, int N, int K)
{
    __shared__ __align__(16) unsigned short tA[2][128 * LDS_PITCH];
    __shared__ __align__(16) unsigned short tB[2][128 * LDS_PITCH];

    const int lane = threadIdx.x & 31;
    const int wave = threadIdx.x >> 5;
    const int mblk = blockIdx.y * 128;
    const int nblk = blockIdx.x * 128;
    const int mw = (wave >> 1) * 64;  // wave tile origin inside block tile
    const int nw = (wave & 1) * 64;

    v8f acc[4][4];
#pragma unroll
    for (int mi = 0; mi < 4; ++mi)
#pragma unroll
        for (int ni = 0; ni < 4; ++ni) { v8f z = {}; acc[mi][ni] = z; }

    const int nsteps = K / 32;
    if (wave == 0) {
        tdm_load_tile(A + (size_t)mblk * K, lds_addr_of(tA[0]), 128, K, M, K);
        tdm_load_tile(W + (size_t)nblk * K, lds_addr_of(tB[0]), 128, K, N, K);
    }

    for (int i = 0; i < nsteps; ++i) {
        const int cur = i & 1;
        if (wave == 0) {
            if (i + 1 < nsteps) {
                const size_t k1 = (size_t)(i + 1) * 32;
                tdm_load_tile(A + (size_t)mblk * K + k1, lds_addr_of(tA[cur ^ 1]),
                              128, K, M, K);
                tdm_load_tile(W + (size_t)nblk * K + k1, lds_addr_of(tB[cur ^ 1]),
                              128, K, N, K);
                __builtin_amdgcn_s_wait_tensorcnt(2);  // current pair complete
            } else {
                __builtin_amdgcn_s_wait_tensorcnt(0);
            }
        }
        __syncthreads();

        const unsigned short* Ab = &tA[cur][0];
        const unsigned short* Bb = &tB[cur][0];
        Frag af[4], bf[4];
#pragma unroll
        for (int mi = 0; mi < 4; ++mi)
            load_afrag(af[mi], Ab + (size_t)(mw + mi * 16) * LDS_PITCH, LDS_PITCH, lane);
#pragma unroll
        for (int ni = 0; ni < 4; ++ni)
            load_bfrag(bf[ni], Bb + (size_t)(nw + ni * 16) * LDS_PITCH, LDS_PITCH, lane);
#pragma unroll
        for (int mi = 0; mi < 4; ++mi)
#pragma unroll
            for (int ni = 0; ni < 4; ++ni)
                acc[mi][ni] = bwmma(af[mi], bf[ni], acc[mi][ni]);

        __syncthreads();
    }

    const int mloc = ((lane >> 4) & 1) * 8;
    const int nloc = lane & 15;
#pragma unroll
    for (int mi = 0; mi < 4; ++mi)
#pragma unroll
        for (int ni = 0; ni < 4; ++ni)
#pragma unroll
            for (int j = 0; j < 8; ++j) {
                int m = mblk + mw + mi * 16 + mloc + j;
                int n = nblk + nw + ni * 16 + nloc;
                float val = acc[mi][ni][j];
                if (MODE == 0)
                    ((unsigned short*)Cout)[(size_t)m * N + n] = f2bf(val);
                else if (MODE == 1)
                    ((unsigned short*)Cout)[(size_t)n * M + m] = f2bf(val);
                else
                    ((float*)Cout)[(size_t)m * N + n] = val;
            }
}

#else  // !USE_TDM ---------------------------------------------------------
// Fallback: fragments loaded directly from global (L2-resident operands).
template <int MODE>
__global__ __launch_bounds__(128) void gemm_bf16_kernel(
    const unsigned short* __restrict__ A,
    const unsigned short* __restrict__ W,
    void* __restrict__ Cout,
    int M, int N, int K)
{
    const int lane = threadIdx.x & 31;
    const int wave = threadIdx.x >> 5;
    const int mw = blockIdx.y * 128 + (wave >> 1) * 64;
    const int nw = blockIdx.x * 128 + (wave & 1) * 64;

    v8f acc[4][4];
#pragma unroll
    for (int mi = 0; mi < 4; ++mi)
#pragma unroll
        for (int ni = 0; ni < 4; ++ni) { v8f z = {}; acc[mi][ni] = z; }

    for (int k0 = 0; k0 < K; k0 += 32) {
        Frag af[4], bf[4];
#pragma unroll
        for (int mi = 0; mi < 4; ++mi)
            load_afrag(af[mi], A + (size_t)(mw + mi * 16) * K + k0, K, lane);
#pragma unroll
        for (int ni = 0; ni < 4; ++ni)
            load_bfrag(bf[ni], W + (size_t)(nw + ni * 16) * K + k0, K, lane);
#pragma unroll
        for (int mi = 0; mi < 4; ++mi)
#pragma unroll
            for (int ni = 0; ni < 4; ++ni)
                acc[mi][ni] = bwmma(af[mi], bf[ni], acc[mi][ni]);
    }

    const int mloc = ((lane >> 4) & 1) * 8;
    const int nloc = lane & 15;
#pragma unroll
    for (int mi = 0; mi < 4; ++mi)
#pragma unroll
        for (int ni = 0; ni < 4; ++ni)
#pragma unroll
            for (int j = 0; j < 8; ++j) {
                int m = mw + mi * 16 + mloc + j;
                int n = nw + ni * 16 + nloc;
                float val = acc[mi][ni][j];
                if (MODE == 0)
                    ((unsigned short*)Cout)[(size_t)m * N + n] = f2bf(val);
                else if (MODE == 1)
                    ((unsigned short*)Cout)[(size_t)n * M + m] = f2bf(val);
                else
                    ((float*)Cout)[(size_t)m * N + n] = val;
            }
}
#endif  // USE_TDM

// ---------------------------------------------------------------------------
// Causal flash attention, bf16 WMMA, online softmax.
// Q,K: [B*S, E] bf16.  Vt: [E, B*S] bf16 (pre-transposed).  Aout: [B*S, E] bf16.
// Block: 128 threads = 4 waves; wave handles 16 query rows; grid (S/64, H, B).
// ---------------------------------------------------------------------------
__global__ __launch_bounds__(128) void attn_kernel(
    const unsigned short* __restrict__ Q,
    const unsigned short* __restrict__ Km,
    const unsigned short* __restrict__ Vt,
    unsigned short* __restrict__ Aout,
    int S, int E, int HD, int MR /* = B*S, Vt row stride */)
{
    __shared__ __align__(16) unsigned short ldsP[4][16][40];  // per-wave P tile

    const int lane = threadIdx.x & 31;
    const int wave = threadIdx.x >> 5;
    const int h = blockIdx.y;
    const int b = blockIdx.z;
    const int qb = (blockIdx.x * 4 + wave) * 16;   // query row base (in-sequence)
    const size_t rowBase = (size_t)b * S;
    const int hc = h * HD;

    const int mloc = ((lane >> 4) & 1) * 8;
    const int nloc = lane & 15;
    const float sc = 0.125f;                       // 1/sqrt(64)
    const float NEG = -1e30f;

    float run_m[8], run_l[8];
    v8f o[4];
#pragma unroll
    for (int j = 0; j < 8; ++j) { run_m[j] = NEG; run_l[j] = 0.0f; }
#pragma unroll
    for (int c = 0; c < 4; ++c) { v8f z = {}; o[c] = z; }

    const unsigned short* Qbase = Q + (rowBase + qb) * E + hc;
    Frag qf[2];
    load_afrag(qf[0], Qbase + 0, E, lane);
    load_afrag(qf[1], Qbase + 32, E, lane);

    for (int kb = 0; kb <= qb + 15; kb += 32) {    // wave-uniform causal bound
        v8f s0 = {}, s1 = {};
        const unsigned short* Kbase = Km + (rowBase + kb) * E + hc;
#pragma unroll
        for (int kk = 0; kk < 2; ++kk) {
            Frag b0, b1;
            load_bfrag(b0, Kbase + kk * 32, E, lane);
            load_bfrag(b1, Kbase + (size_t)16 * E + kk * 32, E, lane);
            s0 = bwmma(qf[kk], b0, s0);
            s1 = bwmma(qf[kk], b1, s1);
        }

#pragma unroll
        for (int j = 0; j < 8; ++j) {
            int mg = qb + mloc + j;
            int n0 = kb + nloc;
            int n1 = kb + 16 + nloc;
            float v0 = (n0 <= mg) ? s0[j] * sc : NEG;
            float v1 = (n1 <= mg) ? s1[j] * sc : NEG;

            float tmax = fmaxf(v0, v1);
#pragma unroll
            for (int off = 1; off < 16; off <<= 1)
                tmax = fmaxf(tmax, __shfl_xor(tmax, off, 16));

            float nm = fmaxf(run_m[j], tmax);
            float alpha = __expf(run_m[j] - nm);
            float p0 = __expf(v0 - nm);
            float p1 = __expf(v1 - nm);
            float psum = p0 + p1;
#pragma unroll
            for (int off = 1; off < 16; off <<= 1)
                psum += __shfl_xor(psum, off, 16);

            run_l[j] = run_l[j] * alpha + psum;
            run_m[j] = nm;
#pragma unroll
            for (int c = 0; c < 4; ++c) o[c][j] *= alpha;

            ldsP[wave][mloc + j][nloc]      = f2bf(p0);
            ldsP[wave][mloc + j][nloc + 16] = f2bf(p1);
        }

        Frag pf;
        {
            const unsigned short* p = &ldsP[wave][lane & 15][((lane >> 4) & 1) * 8];
            pf.q[0] = *reinterpret_cast<const u32x4*>(p);
            pf.q[1] = *reinterpret_cast<const u32x4*>(p + 16);
        }
#pragma unroll
        for (int c = 0; c < 4; ++c) {
            Frag vf;
            load_bfrag(vf, Vt + (size_t)(hc + c * 16) * MR + rowBase + kb,
                       (size_t)MR, lane);
            o[c] = bwmma(pf, vf, o[c]);
        }
    }

#pragma unroll
    for (int c = 0; c < 4; ++c)
#pragma unroll
        for (int j = 0; j < 8; ++j) {
            float val = o[c][j] / run_l[j];
            Aout[(rowBase + qb + mloc + j) * E + hc + c * 16 + nloc] = f2bf(val);
        }
}

// ---------------------------------------------------------------------------
// Launcher
// ---------------------------------------------------------------------------
extern "C" void kernel_launch(void* const* d_in, const int* in_sizes, int n_in,
                              void* d_out, int out_size, void* d_ws, size_t ws_size,
                              hipStream_t stream) {
    const float* x  = (const float*)d_in[0];
    const float* Wq = (const float*)d_in[1];
    const float* Wk = (const float*)d_in[2];
    const float* Wv = (const float*)d_in[3];
    const float* Wo = (const float*)d_in[4];

    const int B = 4, S = 2048, E = 1024, H = 16, HD = 64;
    const int M = B * S;  // 8192

    char* ws = (char*)d_ws;
    const size_t MB = 1u << 20;
    unsigned short* xb   = (unsigned short*)(ws + 0 * MB);   // 16 MB  x bf16 [M,E]
    unsigned short* wqb  = (unsigned short*)(ws + 16 * MB);  //  2 MB
    unsigned short* wkb  = (unsigned short*)(ws + 18 * MB);  //  2 MB
    unsigned short* wvb  = (unsigned short*)(ws + 20 * MB);  //  2 MB
    unsigned short* wob  = (unsigned short*)(ws + 22 * MB);  //  2 MB
    unsigned short* Qb   = (unsigned short*)(ws + 24 * MB);  // 16 MB  [M,E]
    unsigned short* Kb   = (unsigned short*)(ws + 40 * MB);  // 16 MB  [M,E]
    unsigned short* Vtb  = (unsigned short*)(ws + 56 * MB);  // 16 MB  [E,M]
    unsigned short* attb = (unsigned short*)(ws + 72 * MB);  // 16 MB  [M,E]

    {
        int n = M * E;
        cast_bf16_kernel<<<(n + 255) / 256, 256, 0, stream>>>(x, xb, n);
        int nw = E * E;
        cast_bf16_kernel<<<(nw + 255) / 256, 256, 0, stream>>>(Wq, wqb, nw);
        cast_bf16_kernel<<<(nw + 255) / 256, 256, 0, stream>>>(Wk, wkb, nw);
        cast_bf16_kernel<<<(nw + 255) / 256, 256, 0, stream>>>(Wv, wvb, nw);
        cast_bf16_kernel<<<(nw + 255) / 256, 256, 0, stream>>>(Wo, wob, nw);
    }

    dim3 gg(E / 128, M / 128);  // (8, 64)
    gemm_bf16_kernel<0><<<gg, 128, 0, stream>>>(xb, wqb, Qb,  M, E, E);
    gemm_bf16_kernel<0><<<gg, 128, 0, stream>>>(xb, wkb, Kb,  M, E, E);
    gemm_bf16_kernel<1><<<gg, 128, 0, stream>>>(xb, wvb, Vtb, M, E, E);

    dim3 ga(S / 64, H, B);  // (32, 16, 4)
    attn_kernel<<<ga, 128, 0, stream>>>(Qb, Kb, Vtb, attb, S, E, HD, M);

    gemm_bf16_kernel<2><<<gg, 128, 0, stream>>>(attb, wob, d_out, M, E, E);
}
